// ProtHGT_48687749267607
// MI455X (gfx1250) — matrix operand order repaired and checked by reading, not compiled, not run on hardware
//
#include <hip/hip_runtime.h>
#include <cmath>

// ---------------------------------------------------------------------------
// MI455X (gfx1250, wave32) implementation of the ProtHGT reference.
//
// Input index map (setup_inputs dict order; params expanded in make_params
// insertion order):
//   0 x_protein [NP,1024]   1 x_go [NG,768]
//   2 ppi_src 3 ppi_dst 4 ann_src 5 ann_dst 6 pair_row 7 pair_col
//   8.. params (62 tensors):
//     +0 emb_Protein_w +1 emb_Protein_b +2 emb_GO_w +3 emb_GO_b
//     per layer l (base 4+27l):
//       Protein: +0 k_w +1 k_b +2 q_w +3 q_b +4 v_w +5 v_b +6 a_w +7 a_b +8 skip
//       GO:      +9..+17 (same order)
//       rels:    +18 arel_ppi +19 mrel_ppi +20 prel_ppi
//                +21 arel_ann +22 mrel_ann +23 prel_ann
//                +24 arel_rev +25 mrel_rev +26 prel_rev
//     +58 mlp0_w +59 mlp0_b +60 mlp1_w +61 mlp1_b
//   n_in-2 rev_src   n_in-1 rev_dst
// Output: preds[P] ++ x_Protein[NP*128] ++ x_GO[NG*128]  (float32)
//
// Strategy: weights are pre-transposed + pre-converted to bf16 (WT[n][k]) once;
// the WMMA GEMM streams f32 activations (converted in-regs) against WT read
// straight from global (L2/WGP$-resident on MI455X's 192MB L2). No LDS needed.
// ---------------------------------------------------------------------------

typedef __attribute__((ext_vector_type(16))) __bf16    v16bf;
typedef __attribute__((ext_vector_type(2)))  __bf16    v2bf;
typedef __attribute__((ext_vector_type(8)))  float     v8f;
typedef __attribute__((ext_vector_type(8)))  unsigned  v8u;
typedef __attribute__((ext_vector_type(4)))  unsigned  v4u;

struct BFragBits { v4u lo, hi; };   // 32 bytes == v16bf

// f32 -> bf16 via clang native conversion (hardware cvt when available)
static __device__ __forceinline__ unsigned short bfh(float f) {
  return __builtin_bit_cast(unsigned short, (__bf16)f);
}
// pack two floats into one dword of bf16 (f0 -> lo, f1 -> hi)
static __device__ __forceinline__ unsigned pkbf(float f0, float f1) {
  v2bf r = { (__bf16)f0, (__bf16)f1 };
  return __builtin_bit_cast(unsigned, r);
}

// monotone float->uint encoding so atomicMax(uint) == float max
static __device__ __forceinline__ unsigned fenc(float f) {
  unsigned u = __float_as_uint(f);
  return (u & 0x80000000u) ? ~u : (u | 0x80000000u);
}
static __device__ __forceinline__ float fdec(unsigned u) {
  unsigned b = (u & 0x80000000u) ? (u ^ 0x80000000u) : ~u;
  return __uint_as_float(b);
}

// assemble A fragment (v16bf) from 16B-aligned f32 row: K halves {ko..ko+15}
static __device__ __forceinline__ v16bf load_a_frag(const float* __restrict__ Arow, int ko) {
  const float4 a0 = *(const float4*)(Arow + ko);
  const float4 a1 = *(const float4*)(Arow + ko + 4);
  const float4 a2 = *(const float4*)(Arow + ko + 8);
  const float4 a3 = *(const float4*)(Arow + ko + 12);
  v8u au;
  au[0] = pkbf(a0.x, a0.y); au[1] = pkbf(a0.z, a0.w);
  au[2] = pkbf(a1.x, a1.y); au[3] = pkbf(a1.z, a1.w);
  au[4] = pkbf(a2.x, a2.y); au[5] = pkbf(a2.z, a2.w);
  au[6] = pkbf(a3.x, a3.y); au[7] = pkbf(a3.z, a3.w);
  return __builtin_bit_cast(v16bf, au);
}

// B fragment: 16 contiguous bf16 halves at a 16B-aligned pointer (2x b128)
static __device__ __forceinline__ v16bf load_b_frag(const unsigned short* __restrict__ bp) {
  BFragBits bb;
  bb.lo = *(const v4u*)(bp);
  bb.hi = *(const v4u*)(bp + 8);
  return __builtin_bit_cast(v16bf, bb);
}

// ---------------------------------------------------------------------------
// One-shot weight transpose+convert: W[K,N] f32 -> WT[N][K] bf16 (row stride K)
// ---------------------------------------------------------------------------
__global__ __launch_bounds__(256)
void wt_convert(const float* __restrict__ W, unsigned* __restrict__ WT,
                int K, int N)
{
  int t = blockIdx.x * 256 + threadIdx.x;     // dword index: n*(K/2) + kp
  int kd = K >> 1;
  if (t >= N * kd) return;
  int n  = t / kd;
  int kp = t - n * kd;
  float f0 = W[(size_t)(2 * kp)     * N + n];
  float f1 = W[(size_t)(2 * kp + 1) * N + n];
  WT[t] = pkbf(f0, f1);
}

// ---------------------------------------------------------------------------
// Generic WMMA GEMM:  C[M,N] = act(A[M,K](f32) * WT[N,K](bf16) + bias)
// 64x64 block tile, 4 waves (128 thr); wave owns a 16-row strip, 4 N-tiles.
// A: float4 global loads + in-reg bf16 convert.  B: 2x global b128 per tile.
// act: 0=none, 1=relu, 2=skip-blend: out = g*o + (1-g)*Xprev, g=sigmoid(*skipGate)
// ---------------------------------------------------------------------------
__global__ __launch_bounds__(128)
void gemm_bf16t_kernel(const float* __restrict__ A, int lda,
                       const unsigned short* __restrict__ WT,
                       const float* __restrict__ bias,
                       float* __restrict__ C, int ldc,
                       int M, int N, int K, int act,
                       const float* __restrict__ skipGate,
                       const float* __restrict__ Xprev, int ldx)
{
  const int tid  = threadIdx.x;
  const int wave = tid >> 5;
  const int lane = tid & 31;
  const int li   = lane & 15;   // position within half-wave
  const int hs   = lane >> 4;   // half select

  const int bm = blockIdx.x * 64;
  const int bn = blockIdx.y * 64;

  int rowA = bm + wave * 16 + li;          // A fragment: both halves map rows 0..15
  if (rowA >= M) rowA = M - 1;             // clamp loads; stores predicated below
  const float* Arow = A + (size_t)rowA * lda;

  // per-lane B fragment pointers: col = bn + nt*16 + li, halves {k0+hs*16 ..}
  const unsigned short* bp0 = WT + (size_t)(bn + li) * K + hs * 16;
  const unsigned short* bp1 = bp0 + (size_t)16 * K;
  const unsigned short* bp2 = bp1 + (size_t)16 * K;
  const unsigned short* bp3 = bp2 + (size_t)16 * K;

  v8f zero = {0.f,0.f,0.f,0.f,0.f,0.f,0.f,0.f};
  v8f acc[4];
#pragma unroll
  for (int i = 0; i < 4; ++i) acc[i] = zero;

  for (int k0 = 0; k0 < K; k0 += 32) {
    if (k0 + 32 < K) __builtin_prefetch(Arow + k0 + 32, 0, 1);

    v16bf af = load_a_frag(Arow, k0 + hs * 16);
    v16bf b0 = load_b_frag(bp0);
    v16bf b1 = load_b_frag(bp1);
    v16bf b2 = load_b_frag(bp2);
    v16bf b3 = load_b_frag(bp3);

    acc[0] = __builtin_amdgcn_wmma_f32_16x16x32_bf16(false, af, false, b0, (short)0, acc[0], false, false);
    acc[1] = __builtin_amdgcn_wmma_f32_16x16x32_bf16(false, af, false, b1, (short)0, acc[1], false, false);
    acc[2] = __builtin_amdgcn_wmma_f32_16x16x32_bf16(false, af, false, b2, (short)0, acc[2], false, false);
    acc[3] = __builtin_amdgcn_wmma_f32_16x16x32_bf16(false, af, false, b3, (short)0, acc[3], false, false);

    bp0 += 32; bp1 += 32; bp2 += 32; bp3 += 32;
  }

  // epilogue
  float g = 1.f, omg = 0.f;
  if (act == 2) { float s = skipGate[0]; g = 1.f / (1.f + __expf(-s)); omg = 1.f - g; }
  const int row0 = bm + wave * 16 + (hs ? 8 : 0);   // C: vgpr r -> row r (half0) / r+8 (half1)
#pragma unroll
  for (int nt = 0; nt < 4; ++nt) {
    int col = bn + nt * 16 + li;
    float bv = bias ? bias[col] : 0.f;
#pragma unroll
    for (int r = 0; r < 8; ++r) {
      int row = row0 + r;
      if (row < M) {
        float o = acc[nt][r] + bv;
        if (act == 1)      o = fmaxf(o, 0.f);
        else if (act == 2) o = g * o + omg * Xprev[(size_t)row * ldx + col];
        C[(size_t)row * ldc + col] = o;
      }
    }
  }
}

// ---------------------------------------------------------------------------
// Fold relation matrix into projection weights, emit TRANSPOSED bf16 directly:
//   WT[(base+j), i] = bf16( sum_d W[i, h*32+d] * rel[h, d, j%32] )  (h = j/32)
//   bp[base+j]      = sum_d b[h*32+d] * rel[h, d, j%32]
// WT row stride = 128 halves.  grid = 129 blocks, block = 128 threads (j)
// ---------------------------------------------------------------------------
__global__ __launch_bounds__(128)
void pack_weight_t(const float* __restrict__ W, const float* __restrict__ b,
                   const float* __restrict__ rel,
                   unsigned short* __restrict__ WT, float* __restrict__ bp,
                   int baseCol)
{
  int j  = threadIdx.x;
  int h  = j >> 5, jd = j & 31;
  int i  = blockIdx.x;
  if (i < 128) {
    const float* wrow = W + i * 128 + h * 32;
    float acc = 0.f;
#pragma unroll
    for (int d = 0; d < 32; ++d) acc += wrow[d] * rel[(h * 32 + d) * 32 + jd];
    WT[(size_t)(baseCol + j) * 128 + i] = bfh(acc);
  } else {
    float acc = 0.f;
#pragma unroll
    for (int d = 0; d < 32; ++d) acc += b[h * 32 + d] * rel[(h * 32 + d) * 32 + jd];
    bp[baseCol + j] = acc;
  }
}

// ---------------------------------------------------------------------------
// Edge kernels (segment softmax + scatter).  1 thread per (edge, head).
// q/k rows are 512B-aligned -> float4 loads (8 b128 per dot instead of 32 b32)
// ---------------------------------------------------------------------------
__global__ __launch_bounds__(256)
void edge_alpha_max(const int* __restrict__ src, const int* __restrict__ dst, int E,
                    const float* __restrict__ q, int ldq,
                    const float* __restrict__ k, int ldk,
                    const float* __restrict__ prel, float invSqrtD,
                    float* __restrict__ alpha, unsigned* __restrict__ amax)
{
  int t = blockIdx.x * 256 + threadIdx.x;
  if (t >= E * 4) return;
  int e = t >> 2, h = t & 3;
  int s = src[e], d = dst[e];
  const float4* qp = (const float4*)(q + (size_t)d * ldq + h * 32);
  const float4* kp = (const float4*)(k + (size_t)s * ldk + h * 32);
  float acc = 0.f;
#pragma unroll
  for (int i = 0; i < 8; ++i) {
    float4 qv = qp[i], kv = kp[i];
    acc += qv.x * kv.x + qv.y * kv.y + qv.z * kv.z + qv.w * kv.w;
  }
  acc *= prel[h] * invSqrtD;
  alpha[t] = acc;
  atomicMax(amax + (size_t)d * 4 + h, fenc(acc));
}

__global__ __launch_bounds__(256)
void edge_exp_den(const int* __restrict__ dst, int E,
                  const unsigned* __restrict__ amax,
                  float* __restrict__ alpha, float* __restrict__ den)
{
  int t = blockIdx.x * 256 + threadIdx.x;
  if (t >= E * 4) return;
  int e = t >> 2, h = t & 3;
  int d = dst[e];
  float m = fdec(amax[(size_t)d * 4 + h]);
  if (!(m > -3.0e38f && m < 3.0e38f)) m = 0.f;   // match reference isfinite clamp
  float ex = __expf(alpha[t] - m);
  alpha[t] = ex;
  atomicAdd(den + (size_t)d * 4 + h, ex);
}

// 32 threads per edge; thread j covers cols {j, 32+j, 64+j, 96+j}
__global__ __launch_bounds__(256)
void edge_scatter(const int* __restrict__ src, const int* __restrict__ dst, int E,
                  const float* __restrict__ v, int ldv,
                  const float* __restrict__ ebuf, const float* __restrict__ den,
                  float* __restrict__ agg)
{
  long t = (long)blockIdx.x * 256 + threadIdx.x;
  if (t >= (long)E * 32) return;
  int e = (int)(t >> 5), j = (int)(t & 31);
  int s = src[e], d = dst[e];
  const float* vp = v + (size_t)s * ldv;
  float* ap = agg + (size_t)d * 128;
#pragma unroll
  for (int h = 0; h < 4; ++h) {
    int c = h * 32 + j;
    float coef = ebuf[e * 4 + h] / (den[(size_t)d * 4 + h] + 1e-16f);
    atomicAdd(ap + c, vp[c] * coef);
  }
}

// ---------------------------------------------------------------------------
// elementwise helpers
// ---------------------------------------------------------------------------
__global__ __launch_bounds__(256)
void fill_f32(float* __restrict__ p, long n, float v) {
  long i = (long)blockIdx.x * 256 + threadIdx.x;
  if (i < n) p[i] = v;
}
__global__ __launch_bounds__(256)
void fill_u32(unsigned* __restrict__ p, long n, unsigned v) {
  long i = (long)blockIdx.x * 256 + threadIdx.x;
  if (i < n) p[i] = v;
}
__global__ __launch_bounds__(256)
void gelu_kernel(float* __restrict__ p, long n) {
  long i = (long)blockIdx.x * 256 + threadIdx.x;
  if (i < n) {
    float x = p[i];
    p[i] = 0.5f * x * (1.f + erff(x * 0.70710678118654752f));
  }
}

// ---------------------------------------------------------------------------
// Fused pair MLP: preds[p] = relu(concat(xP[row],xG[col]) @ W0 + b0) @ w1 + b1
// One wave per 16 pairs.  W0T is pre-converted bf16 [128 cols][256 K].
// 64 WMMAs, then relu + dot(w1) + cross-lane reduction.
// ---------------------------------------------------------------------------
__global__ __launch_bounds__(32)
void pair_mlp(const int* __restrict__ prow, const int* __restrict__ pcol, int P,
              const float* __restrict__ xP, const float* __restrict__ xG,
              const unsigned short* __restrict__ W0T, const float* __restrict__ b0,
              const float* __restrict__ w1, const float* __restrict__ b1,
              float* __restrict__ preds)
{
  const int lane = threadIdx.x;
  const int li   = lane & 15;
  const int hs   = lane >> 4;
  const int base = blockIdx.x * 16;

  int pr = base + li;
  if (pr >= P) pr = P - 1;
  const int rP = prow[pr];
  const int cG = pcol[pr];

  v8f zero = {0.f,0.f,0.f,0.f,0.f,0.f,0.f,0.f};
  v8f acc[8];
#pragma unroll
  for (int i = 0; i < 8; ++i) acc[i] = zero;

#pragma unroll
  for (int ks = 0; ks < 8; ++ks) {
    const int k0 = ks * 32;
    const float* srcv;
    int cb;
    if (k0 < 128) { srcv = xP + (size_t)rP * 128; cb = k0; }
    else          { srcv = xG + (size_t)cG * 128; cb = k0 - 128; }
    v16bf af = load_a_frag(srcv, cb + hs * 16);

    const int kb = k0 + hs * 16;
#pragma unroll
    for (int nt = 0; nt < 8; ++nt) {
      const unsigned short* bp = W0T + (size_t)(nt * 16 + li) * 256 + kb;
      v16bf bf = load_b_frag(bp);
      acc[nt] = __builtin_amdgcn_wmma_f32_16x16x32_bf16(
          false, af, false, bf, (short)0, acc[nt], false, false);
    }
  }

  // relu + second-layer dot: lane (hs,li) holds cols {nt*16+li} of rows r(+8*hs)
  float rowsum[8];
#pragma unroll
  for (int r = 0; r < 8; ++r) rowsum[r] = 0.f;
#pragma unroll
  for (int nt = 0; nt < 8; ++nt) {
    int col = nt * 16 + li;
    float bb = b0[col], ww = w1[col];
#pragma unroll
    for (int r = 0; r < 8; ++r) {
      float o = fmaxf(acc[nt][r] + bb, 0.f);
      rowsum[r] += o * ww;
    }
  }
  // reduce over the 16 lanes of each half (xor masks < 16 keep halves separate)
#pragma unroll
  for (int r = 0; r < 8; ++r) {
    float v = rowsum[r];
    v += __shfl_xor(v, 1, 32);
    v += __shfl_xor(v, 2, 32);
    v += __shfl_xor(v, 4, 32);
    v += __shfl_xor(v, 8, 32);
    rowsum[r] = v;
  }
  if (li == 0) {
    float bb1 = b1[0];
#pragma unroll
    for (int r = 0; r < 8; ++r) {
      int pidx = base + r + (hs ? 8 : 0);
      if (pidx < P) preds[pidx] = rowsum[r] + bb1;
    }
  }
}

// ---------------------------------------------------------------------------
// host orchestration
// ---------------------------------------------------------------------------
extern "C" void kernel_launch(void* const* d_in, const int* in_sizes, int n_in,
                              void* d_out, int out_size, void* d_ws, size_t ws_size,
                              hipStream_t stream)
{
  (void)out_size; (void)ws_size;

  const float* x_protein = (const float*)d_in[0];
  const float* x_go      = (const float*)d_in[1];
  const int* ppi_src  = (const int*)d_in[2];
  const int* ppi_dst  = (const int*)d_in[3];
  const int* ann_src  = (const int*)d_in[4];
  const int* ann_dst  = (const int*)d_in[5];
  const int* pair_row = (const int*)d_in[6];
  const int* pair_col = (const int*)d_in[7];
  const int* rev_src  = (const int*)d_in[n_in - 2];
  const int* rev_dst  = (const int*)d_in[n_in - 1];

  const int NP = in_sizes[0] / 1024;
  const int NG = in_sizes[1] / 768;
  const int E_ppi = in_sizes[2];
  const int E_ann = in_sizes[4];
  const int E_rev = in_sizes[n_in - 2];
  const int Pn = in_sizes[6];

  auto PF = [&](int i) { return (const float*)d_in[8 + i]; };
  const float* embPw = PF(0);  const float* embPb = PF(1);
  const float* embGw = PF(2);  const float* embGb = PF(3);
  const float* mlp0w = PF(58); const float* mlp0b = PF(59);
  const float* mlp1w = PF(60); const float* mlp1b = PF(61);

  // workspace carve-out
  float* ws = (float*)d_ws;
  size_t off = 0;
  auto alloc  = [&](size_t n) { float* r = ws + off; off += n; return r; };
  auto alloch = [&](size_t nhalves) { return (unsigned short*)alloc(nhalves / 2); };
  float* xP0  = alloc((size_t)NP * 128);
  float* xG0  = alloc((size_t)NG * 128);
  float* xP1  = alloc((size_t)NP * 128);
  float* xG1  = alloc((size_t)NG * 128);
  float* qP   = alloc((size_t)NP * 128);
  float* qG   = alloc((size_t)NG * 128);
  float* kvB  = alloc((size_t)NP * 256);   // [Nsrc, 256]: cols 0-127 = K_rel, 128-255 = V_rel
  float* aggP = alloc((size_t)NP * 128);
  float* aggG = alloc((size_t)NG * 128);
  int Emax = E_ppi; if (E_ann > Emax) Emax = E_ann; if (E_rev > Emax) Emax = E_rev;
  float*    alphaB = alloc((size_t)Emax * 4);
  unsigned* amaxB  = (unsigned*)alloc((size_t)NP * 4);
  float*    denB   = alloc((size_t)NP * 4);
  float*    bkv    = alloc(256);
  // pre-converted bf16 weights (transposed, row stride = K)
  unsigned short* embPwT = alloch(128 * 1024);
  unsigned short* embGwT = alloch(128 * 768);
  unsigned short* mlp0T  = alloch(128 * 256);
  unsigned short* qwT[2][2]; unsigned short* awT[2][2];
  for (int l = 0; l < 2; ++l)
    for (int ntp = 0; ntp < 2; ++ntp) {
      qwT[l][ntp] = alloch(128 * 128);
      awT[l][ntp] = alloch(128 * 128);
    }
  unsigned short* WkvT = alloch(256 * 128);   // per-relation folded K|V weight

  float* out_preds = (float*)d_out;
  float* out_xP    = out_preds + Pn;
  float* out_xG    = out_xP + (size_t)NP * 128;

  const float invSqrtD = 0.17677669529663687f;  // 1/sqrt(32)

#define GEMM(Ap, Mv, Kv, ldav, WTp, Nv, bp, Cp, ldcv, actv, sgp, xpp, ldxv)          \
  gemm_bf16t_kernel<<<dim3(((Mv) + 63) / 64, (Nv) / 64), dim3(128), 0, stream>>>(     \
      Ap, ldav, WTp, bp, Cp, ldcv, Mv, Nv, Kv, actv, sgp, xpp, ldxv)
#define WCONV(Wp, Kv, Nv, WTp)                                                        \
  wt_convert<<<dim3((unsigned)(((size_t)(Nv) * ((Kv) / 2) + 255) / 256)), dim3(256),  \
               0, stream>>>(Wp, (unsigned*)(WTp), Kv, Nv)
#define FILLF(p, n) fill_f32<<<dim3((unsigned)(((n) + 255) / 256)), dim3(256), 0, stream>>>(p, (long)(n), 0.f)
#define FILLU(p, n) fill_u32<<<dim3((unsigned)(((n) + 255) / 256)), dim3(256), 0, stream>>>(p, (long)(n), 0u)

  // ---- one-shot weight conversions ----
  WCONV(embPw, 1024, 128, embPwT);
  WCONV(embGw,  768, 128, embGwT);
  WCONV(mlp0w,  256, 128, mlp0T);
  for (int l = 0; l < 2; ++l) {
    const int lb = 4 + 27 * l;
    WCONV(PF(lb + 2),  128, 128, qwT[l][0]);   // q_w Protein
    WCONV(PF(lb + 11), 128, 128, qwT[l][1]);   // q_w GO
    WCONV(PF(lb + 6),  128, 128, awT[l][0]);   // a_w Protein
    WCONV(PF(lb + 15), 128, 128, awT[l][1]);   // a_w GO
  }

  // ---- input embeddings (relu epilogue) ----
  GEMM(x_protein, NP, 1024, 1024, embPwT, 128, embPb, xP0, 128, 1, nullptr, nullptr, 0);
  GEMM(x_go,      NG,  768,  768, embGwT, 128, embGb, xG0, 128, 1, nullptr, nullptr, 0);

  float* xPc = xP0; float* xGc = xG0;   // current features
  float* xPo = xP1; float* xGo = xG1;   // layer output (last layer -> d_out)

  for (int l = 0; l < 2; ++l) {
    const int lb = 4 + 27 * l;
    const float *kwP = PF(lb+0),  *kbP = PF(lb+1),  *qbP = PF(lb+3);
    const float *vwP = PF(lb+4),  *vbP = PF(lb+5),  *abP = PF(lb+7),  *skP = PF(lb+8);
    const float *kwG = PF(lb+9),  *kbG = PF(lb+10), *qbG = PF(lb+12);
    const float *vwG = PF(lb+13), *vbG = PF(lb+14), *abG = PF(lb+16), *skG = PF(lb+17);

    // Q projections (no relation transform)
    GEMM(xPc, NP, 128, 128, qwT[l][0], 128, qbP, qP, 128, 0, nullptr, nullptr, 0);
    GEMM(xGc, NG, 128, 128, qwT[l][1], 128, qbG, qG, 128, 0, nullptr, nullptr, 0);

    FILLF(aggP, (size_t)NP * 128);
    FILLF(aggG, (size_t)NG * 128);

    struct R {
      const int *s, *d; int E;
      const float* Xs; int Ns;
      const float *kw, *kb, *vw, *vb, *arel, *mrel, *prel;
      const float* q; float* agg; int Nd;
    };
    R rels[3] = {
      { ppi_src, ppi_dst, E_ppi, xPc, NP, kwP, kbP, vwP, vbP, PF(lb+18), PF(lb+19), PF(lb+20), qP, aggP, NP },
      { ann_src, ann_dst, E_ann, xPc, NP, kwP, kbP, vwP, vbP, PF(lb+21), PF(lb+22), PF(lb+23), qG, aggG, NG },
      { rev_src, rev_dst, E_rev, xGc, NG, kwG, kbG, vwG, vbG, PF(lb+24), PF(lb+25), PF(lb+26), qP, aggP, NP },
    };

    for (int r = 0; r < 3; ++r) {
      R& e = rels[r];
      // fold arel/mrel into K/V weights (directly transposed bf16), single GEMM
      pack_weight_t<<<dim3(129), dim3(128), 0, stream>>>(e.kw, e.kb, e.arel, WkvT, bkv, 0);
      pack_weight_t<<<dim3(129), dim3(128), 0, stream>>>(e.vw, e.vb, e.mrel, WkvT, bkv, 128);
      GEMM(e.Xs, e.Ns, 128, 128, WkvT, 256, bkv, kvB, 256, 0, nullptr, nullptr, 0);

      FILLU(amaxB, (size_t)e.Nd * 4);
      FILLF(denB,  (size_t)e.Nd * 4);

      long t1 = (long)e.E * 4;
      edge_alpha_max<<<dim3((unsigned)((t1 + 255) / 256)), dim3(256), 0, stream>>>(
          e.s, e.d, e.E, e.q, 128, kvB, 256, e.prel, invSqrtD, alphaB, amaxB);
      edge_exp_den<<<dim3((unsigned)((t1 + 255) / 256)), dim3(256), 0, stream>>>(
          e.d, e.E, amaxB, alphaB, denB);
      long t2 = (long)e.E * 32;
      edge_scatter<<<dim3((unsigned)((t2 + 255) / 256)), dim3(256), 0, stream>>>(
          e.s, e.d, e.E, kvB + 128, 256, alphaB, denB, e.agg);
    }

    long nP = (size_t)NP * 128, nG = (size_t)NG * 128;
    gelu_kernel<<<dim3((unsigned)((nP + 255) / 256)), dim3(256), 0, stream>>>(aggP, nP);
    gelu_kernel<<<dim3((unsigned)((nG + 255) / 256)), dim3(256), 0, stream>>>(aggG, nG);

    // a-projection with sigmoid-gated skip epilogue
    GEMM(aggP, NP, 128, 128, awT[l][0], 128, abP, xPo, 128, 2, skP, xPc, 128);
    GEMM(aggG, NG, 128, 128, awT[l][1], 128, abG, xGo, 128, 2, skG, xGc, 128);

    xPc = xPo; xGc = xGo;
    xPo = out_xP; xGo = out_xG;   // layer 1 writes final features straight to d_out
  }

  // pair scorer (reads final features from d_out regions)
  pair_mlp<<<dim3((unsigned)((Pn + 15) / 16)), dim3(32), 0, stream>>>(
      pair_row, pair_col, Pn, out_xP, out_xG, mlp0T, mlp0b, mlp1w, mlp1b, out_preds);

#undef GEMM
#undef WCONV
#undef FILLF
#undef FILLU
}
